// STAttention_39427799777693
// MI455X (gfx1250) — matrix operand order repaired and checked by reading, not compiled, not run on hardware
//
#include <hip/hip_runtime.h>
#include <hip/hip_bf16.h>
#include <math.h>

// CDNA5 / gfx1250: wave32, WMMA 16x16x32 bf16 -> f32 accum, TDM for x tiles.
typedef __attribute__((ext_vector_type(16))) __bf16        v16bf;
typedef __attribute__((ext_vector_type(8)))  float         v8f;
typedef __attribute__((ext_vector_type(4)))  unsigned int  v4u;
typedef __attribute__((ext_vector_type(8)))  unsigned int  v8u;

#define SCALE_F 0.17677669529663687f  // 32^-0.5 (DIM_HEAD**-0.5, faithful to ref)

// ---------------------------------------------------------------------------
// Prep: convert f32 weights into WMMA B-operand lane layout, bf16, in d_ws.
//   wb1: [kk(8)][tile(6)][lane(32)][e(16)]  for combined W = [Wq | Wk | Wv] (256x96)
//        K = kk*32 + (lane>=16 ? 8 : 0) + (e&7) + (e>=8 ? 16 : 0) ; N = tile*16 + lane%16
//   wob: [tile(16)][lane(32)][e(16)]        for Wo (32x256)
// ---------------------------------------------------------------------------
__global__ void prep_weights(const float* __restrict__ Wq,
                             const float* __restrict__ Wkv,
                             const float* __restrict__ Wo,
                             __bf16* __restrict__ wb1,
                             __bf16* __restrict__ wob) {
  int idx = blockIdx.x * blockDim.x + threadIdx.x;
  if (idx < 24576) {                         // 8*6*32*16
    int e    = idx & 15;
    int lane = (idx >> 4) & 31;
    int tt   = idx >> 9;                     // kk*6 + t, 0..47
    int t    = tt % 6;
    int kk   = tt / 6;
    int K    = kk * 32 + ((lane >> 4) << 3) + (e & 7) + ((e >> 3) << 4);
    int c    = t * 16 + (lane & 15);         // 0..95 in [Wq | Wkv]
    float v  = (c < 32) ? Wq[K * 32 + c] : Wkv[K * 64 + (c - 32)];
    wb1[idx] = (__bf16)v;
  } else if (idx < 32768) {                  // + 16*32*16
    int i2   = idx - 24576;
    int e    = i2 & 15;
    int lane = (i2 >> 4) & 31;
    int t    = i2 >> 9;                      // 0..15
    int K    = ((lane >> 4) << 3) + (e & 7) + ((e >> 3) << 4);       // 0..31
    int N    = t * 16 + (lane & 15);
    wob[i2]  = (__bf16)Wo[K * 256 + N];
  }
}

// ---------------------------------------------------------------------------
// Fused attention: 1 wave = 4 batches (16 rows = one WMMA M tile),
// 2 waves / block -> 8 batches / block. x tile via Tensor Data Mover.
// ---------------------------------------------------------------------------
#define XPITCH 260   // 256 + 4 dwords TDM pad -> conflict-free LDS rows

__global__ __launch_bounds__(64)
void fused_attn(const float* __restrict__ x,
                const __bf16* __restrict__ wb1,
                const __bf16* __restrict__ wob,
                const float* __restrict__ bo,
                const float* __restrict__ pe,      // [1,4,4,4]
                float* __restrict__ yout) {
  __shared__ float xs_st[2][16 * XPITCH];   // per-wave x tile (TDM dest, padded rows)
  __shared__ float qkv_st[2][16 * 100];     // per-wave 16 x [Q(32)|K(32)|V(32)]
  __shared__ float out_st[2][16 * 36];      // per-wave 16 x 32 attention output

  const int lane = threadIdx.x & 31;
  const int wv   = threadIdx.x >> 5;               // wave in block, 0..1
  const int l16  = lane & 15;
  const int hi   = lane >> 4;                      // lane half -> K-offset select
  const int wb   = blockIdx.x * 8 + wv * 4;        // first batch of this wave

  float* xs = xs_st[wv];

  // -------- TDM: DMA 16 rows x 256 f32 of x into LDS, +16B pad per row -----
  {
    const unsigned lds_addr = (unsigned)(size_t)(&xs_st[wv][0]);      // addr[31:0] = LDS offset
    const unsigned long long ga =
        (unsigned long long)(size_t)(x + (size_t)wb * 4 * 256);       // tile start
    v4u g0;
    g0[0] = 1u;                                          // count=1, user mode
    g0[1] = lds_addr;                                    // lds_addr
    g0[2] = (unsigned)(ga & 0xffffffffull);              // global_addr[31:0]
    g0[3] = (unsigned)((ga >> 32) & 0x1ffffffull)        // global_addr[56:32]
            | (2u << 30);                                // type = 2 (image)
    v8u g1;
    g1[0] = (2u << 16)          // data_size = 4B
          | (1u << 20)          // pad_enable
          | (7u << 22)          // pad_interval: 256 DWORDs (= one 1KB row)
          | (3u << 25);         // pad_amount: 4 DWORDs (16B)
    g1[1] = (256u & 0xffffu) << 16;            // tensor_dim0 = 256 (low 16)
    g1[2] = 0u;                                // tensor_dim0 hi | tensor_dim1 lo (524288 & 0xffff = 0)
    g1[3] = 8u | (256u << 16);                 // tensor_dim1 hi (524288>>16) | tile_dim0 = 256
    g1[4] = 16u;                               // tile_dim1 = 16 rows, tile_dim2 = 0
    g1[5] = 256u;                              // tensor_dim0_stride = 256 elements
    g1[6] = 0u;
    g1[7] = 0u;
    // 2-D tensor: groups 2/3 omitted (VADDR2/3 = NULL form).
    asm volatile("tensor_load_to_lds %0, %1" :: "s"(g0), "s"(g1) : "memory");
  }
  __builtin_amdgcn_s_wait_tensorcnt(0);
  asm volatile("" ::: "memory");   // keep LDS reads below the wait

  // ---------------- GEMM1: [16x256] x [256x96] -> Q|K|V (bf16 WMMA) ---------
  v8f acc[6];
#pragma unroll
  for (int t = 0; t < 6; ++t) acc[t] = (v8f){0.f,0.f,0.f,0.f,0.f,0.f,0.f,0.f};

#pragma unroll
  for (int kk = 0; kk < 8; ++kk) {
    const float* xr = xs + l16 * XPITCH + kk * 32 + hi * 8;  // row = lane%16
    const float4 x0 = *(const float4*)(xr);
    const float4 x1 = *(const float4*)(xr + 4);
    const float4 x2 = *(const float4*)(xr + 16);
    const float4 x3 = *(const float4*)(xr + 20);
    const float lo8[8] = {x0.x,x0.y,x0.z,x0.w,x1.x,x1.y,x1.z,x1.w};
    const float hi8[8] = {x2.x,x2.y,x2.z,x2.w,x3.x,x3.y,x3.z,x3.w};
    v16bf a;
#pragma unroll
    for (int e = 0; e < 8; ++e) {
      a[e]     = (__bf16)lo8[e];   // K = base + e
      a[e + 8] = (__bf16)hi8[e];   // K = base + 16 + e
    }
    const __bf16* wp = wb1 + ((size_t)(kk * 6) * 32 + lane) * 16;
#pragma unroll
    for (int t = 0; t < 6; ++t) {
      v16bf b = *(const v16bf*)(wp + t * 512);   // next tile: +32 lanes * 16 elems
      acc[t] = __builtin_amdgcn_wmma_f32_16x16x32_bf16(
                   false, a, false, b, (short)0, acc[t], false, false);
    }
  }

  // Spill Q|K|V to LDS staging (C/D layout: VGPR g -> row g+8*hi, col lane%16)
  float* qkv = qkv_st[wv];
#pragma unroll
  for (int t = 0; t < 6; ++t) {
#pragma unroll
    for (int g = 0; g < 8; ++g)
      qkv[(g + hi * 8) * 100 + t * 16 + l16] = acc[t][g];
  }
  __syncthreads();

  // ---------------- Attention: 16 lanes, one (batch,head) each --------------
  if (lane < 16) {
    const int bl = lane >> 2;                 // local batch 0..3
    const int h  = lane & 3;                  // head 0..3
    const float* rows = qkv + (bl * 4) * 100;
    float qv[4][8], kv[4][8], vv[4][8];
#pragma unroll
    for (int i = 0; i < 4; ++i) {
#pragma unroll
      for (int d = 0; d < 8; ++d) {
        qv[i][d] = rows[i * 100 +      h * 8 + d];
        kv[i][d] = rows[i * 100 + 32 + h * 8 + d];
        vv[i][d] = rows[i * 100 + 64 + h * 8 + d];
      }
    }
    const float* peh = pe + h * 16;           // pos_emb[0][h][i][j]
    float attn[4][4];
#pragma unroll
    for (int i = 0; i < 4; ++i) {
      float s[4];
      float m = -1e30f;
#pragma unroll
      for (int j = 0; j < 4; ++j) {
        float d0 = 0.f;
#pragma unroll
        for (int d = 0; d < 8; ++d) d0 += qv[i][d] * kv[j][d];
        s[j] = d0 * SCALE_F + peh[i * 4 + j];
        m = fmaxf(m, s[j]);
      }
      float sum = 0.f;
#pragma unroll
      for (int j = 0; j < 4; ++j) { s[j] = __expf(s[j] - m); sum += s[j]; }
      const float inv = 1.f / sum;
#pragma unroll
      for (int j = 0; j < 4; ++j) attn[i][j] = s[j] * inv;
    }
    float* orow = out_st[wv] + (bl * 4) * 36;
#pragma unroll
    for (int i = 0; i < 4; ++i) {
#pragma unroll
      for (int d = 0; d < 8; ++d) {
        float o = 0.f;
#pragma unroll
        for (int j = 0; j < 4; ++j) o += attn[i][j] * vv[j][d];
        orow[i * 36 + h * 8 + d] = o;
      }
    }
  }
  __syncthreads();

  // ---------------- GEMM3: [16x32] x [32x256] + bias -> y (bf16 WMMA) -------
  v16bf a2;
  {
    const float* orow = out_st[wv] + l16 * 36 + hi * 8;
#pragma unroll
    for (int e = 0; e < 8; ++e) {
      a2[e]     = (__bf16)orow[e];        // K = hi*8 + e
      a2[e + 8] = (__bf16)orow[16 + e];   // K = hi*8 + 16 + e
    }
  }
  float* yrow0 = yout + (size_t)(wb * 4) * 256;
#pragma unroll
  for (int t = 0; t < 16; ++t) {
    v16bf b = *(const v16bf*)(wob + ((size_t)t * 32 + lane) * 16);
    v8f c = {};
    c = __builtin_amdgcn_wmma_f32_16x16x32_bf16(
            false, a2, false, b, (short)0, c, false, false);
    const int col   = t * 16 + l16;
    const float bia = bo[col];
#pragma unroll
    for (int g = 0; g < 8; ++g)
      yrow0[(size_t)(g + hi * 8) * 256 + col] = c[g] + bia;
  }
}

// ---------------------------------------------------------------------------
extern "C" void kernel_launch(void* const* d_in, const int* in_sizes, int n_in,
                              void* d_out, int out_size, void* d_ws, size_t ws_size,
                              hipStream_t stream) {
  const float* x   = (const float*)d_in[0];
  const float* Wq  = (const float*)d_in[1];
  const float* Wkv = (const float*)d_in[2];
  const float* Wo  = (const float*)d_in[3];
  const float* bo  = (const float*)d_in[4];
  const float* pe  = (const float*)d_in[5];

  __bf16* wb1 = (__bf16*)d_ws;            // 24576 bf16 = 48 KB
  __bf16* wob = wb1 + 24576;              //  8192 bf16 = 16 KB

  prep_weights<<<128, 256, 0, stream>>>(Wq, Wkv, Wo, wb1, wob);

  const int B      = in_sizes[0] / (4 * 256);   // 131072
  const int blocks = B / 8;                     // 8 batches per 64-thread block
  fused_attn<<<blocks, 64, 0, stream>>>(x, wb1, wob, bo, pe, (float*)d_out);
}
